// PhysioGAT_64871186038940
// MI455X (gfx1250) — matrix operand order repaired and checked by reading, not compiled, not run on hardware
//
#include <hip/hip_runtime.h>
#include <hip/hip_bf16.h>

#define NN 30000
#define NE 480000
#define NG 64
#define ET (NE + NN)   // edges + self loops = 510000

typedef _Float16 v16h __attribute__((ext_vector_type(16)));
typedef _Float16 v8h  __attribute__((ext_vector_type(8)));
typedef float    v8f  __attribute__((ext_vector_type(8)));

// ---------------- helpers ----------------

__device__ __forceinline__ void atomicMaxF32(float* addr, float val) {
  // IEEE ordering trick; works for mixed signs when initialized to -FLT_MAX.
  if (val >= 0.0f) atomicMax(reinterpret_cast<int*>(addr), __float_as_int(val));
  else             atomicMin(reinterpret_cast<unsigned int*>(addr), __float_as_uint(val));
}

__device__ __forceinline__ void edge_sd(const int* __restrict__ ei, int e, int& s, int& d) {
  if (e < NE) { s = ei[e]; d = ei[NE + e]; }
  else        { s = e - NE; d = s; }        // self-loop
}

// ---------------- init ----------------

__global__ void k_fill(float* __restrict__ p, float v, int n) {
  int i = blockIdx.x * blockDim.x + threadIdx.x;
  if (i < n) p[i] = v;
}

// ---------------- layer building blocks ----------------

// h1[n, j] = sum_k x[n,k] * W1[k,j],  K = 3
__global__ void k_gemm1(const float* __restrict__ x, const float* __restrict__ W1,
                        float* __restrict__ h1) {
  int idx = blockIdx.x * blockDim.x + threadIdx.x;
  if (idx >= NN * 512) return;
  int n = idx >> 9, j = idx & 511;
  h1[idx] = x[n*3 + 0] * W1[j] + x[n*3 + 1] * W1[512 + j] + x[n*3 + 2] * W1[1024 + j];
}

// a_s[n,hd] = sum_c h[n, hd*C + c] * aw[hd, c];  a_d likewise
__global__ void k_attdot(const float* __restrict__ h,
                         const float* __restrict__ aw, const float* __restrict__ dw,
                         float* __restrict__ a_s, float* __restrict__ a_d,
                         int H, int C) {
  int idx = blockIdx.x * blockDim.x + threadIdx.x;
  if (idx >= NN * H) return;
  int n = idx / H, hd = idx % H;
  const float* hp = h + (size_t)(n * H + hd) * C;
  const float* ap = aw + hd * C;
  const float* dp = dw + hd * C;
  float s = 0.f, d = 0.f;
  for (int c = 0; c < C; ++c) { float v = hp[c]; s += v * ap[c]; d += v * dp[c]; }
  a_s[idx] = s; a_d[idx] = d;
}

template<int H>
__global__ void k_edge_logit(const int* __restrict__ ei,
                             const float* __restrict__ a_s, const float* __restrict__ a_d,
                             float* __restrict__ elog, float* __restrict__ m) {
  int idx = blockIdx.x * blockDim.x + threadIdx.x;
  if (idx >= ET * H) return;
  int e = idx / H, hd = idx % H;
  int s, d; edge_sd(ei, e, s, d);
  float v = a_s[s * H + hd] + a_d[d * H + hd];
  v = (v >= 0.f) ? v : 0.2f * v;            // leaky relu
  elog[idx] = v;
  atomicMaxF32(&m[d * H + hd], v);
}

template<int H>
__global__ void k_edge_exp(const int* __restrict__ ei,
                           float* __restrict__ elog, const float* __restrict__ m,
                           float* __restrict__ denom) {
  int idx = blockIdx.x * blockDim.x + threadIdx.x;
  if (idx >= ET * H) return;
  int e = idx / H, hd = idx % H;
  int s, d; edge_sd(ei, e, s, d);
  float ex = __expf(elog[idx] - m[d * H + hd]);
  elog[idx] = ex;
  atomicAdd(&denom[d * H + hd], ex);
}

// out[dst, :] += (ex/denom[dst]) * h[src, :]; 4 channels per thread
template<int H, int C>
__global__ void k_edge_scatter(const int* __restrict__ ei,
                               const float* __restrict__ ex, const float* __restrict__ denom,
                               const float* __restrict__ h, float* __restrict__ out) {
  const int F   = H * C;
  const int TPE = F / 4;
  int idx = blockIdx.x * blockDim.x + threadIdx.x;
  if (idx >= ET * TPE) return;
  int e = idx / TPE, q = idx % TPE;
  int cbase = q * 4;
  int hd = cbase / C;
  int s, d; edge_sd(ei, e, s, d);
  float alpha = ex[e * H + hd] / denom[d * H + hd];
  const float4 hv = *reinterpret_cast<const float4*>(h + (size_t)s * F + cbase);
  float* op = out + (size_t)d * F + cbase;
  atomicAdd(op + 0, alpha * hv.x);
  atomicAdd(op + 1, alpha * hv.y);
  atomicAdd(op + 2, alpha * hv.z);
  atomicAdd(op + 3, alpha * hv.w);
}

// x1 = relu(acc + b1); also emit f16 copy for the WMMA GEMM
__global__ void k_bias_relu_f16(float* __restrict__ x1, const float* __restrict__ b,
                                _Float16* __restrict__ x1h) {
  int idx = blockIdx.x * blockDim.x + threadIdx.x;
  if (idx >= NN * 512) return;
  float v = x1[idx] + b[idx & 511];
  v = v > 0.f ? v : 0.f;
  x1[idx] = v;
  x1h[idx] = (_Float16)v;
}

// W2t[n, k] = (f16) W2[k, n]   (W2: [512,128] -> W2t: [128,512])
__global__ void k_w2t(const float* __restrict__ W2, _Float16* __restrict__ W2t) {
  int idx = blockIdx.x * blockDim.x + threadIdx.x;
  if (idx >= 512 * 128) return;
  int k = idx >> 7, n = idx & 127;
  W2t[n * 512 + k] = (_Float16)W2[idx];
}

// h2[30000,128] = x1h[30000,512] @ W2[512,128], f16 inputs, f32 accum via WMMA.
// One 16x16 output tile per wave32; K-loop of 16 x v_wmma_f32_16x16x32_f16.
__global__ __launch_bounds__(256)
void k_gemm2_wmma(const _Float16* __restrict__ A, const _Float16* __restrict__ Bt,
                  float* __restrict__ C) {
  int wave = threadIdx.x >> 5;
  int tile = blockIdx.x * 8 + wave;          // 1875*8 = 15000 tiles exactly
  int tm = tile >> 3;                         // row tile (0..1874)
  int tn = tile & 7;                          // col tile (0..7)
  int lane = threadIdx.x & 31;
  int r = lane & 15;                          // M (for A) / N (for B)
  int hgrp = lane >> 4;                       // lane half-group
  const _Float16* arow = A  + (size_t)(tm * 16 + r) * 512;
  const _Float16* brow = Bt + (size_t)(tn * 16 + r) * 512;
  v8f c = {};
#pragma unroll 4
  for (int kb = 0; kb < 512; kb += 32) {
    // A 16x32: elem i -> K = kb + h*16 + i
    // B 32x16 (col-major via pre-transposed W2t): elem i -> K = kb + h*16 + i
    v16h b = *reinterpret_cast<const v16h*>(brow + kb + hgrp * 16);
    v16h a = *reinterpret_cast<const v16h*>(arow + kb + hgrp * 16);
    c = __builtin_amdgcn_wmma_f32_16x16x32_f16(false, a, false, b, (short)0, c,
                                               false, false);
  }
  float* crow = C + (size_t)(tm * 16 + hgrp * 8) * 128 + tn * 16 + r;
#pragma unroll
  for (int v = 0; v < 8; ++v) crow[v * 128] = c[v];   // VGPR v -> row M = v + h*8
}

// x2 = relu(out2 + b2); h = x1[:, :128] + x2; pooled sums += h; counts += 1
__global__ void k_skip_pool(const float* __restrict__ x1, const float* __restrict__ out2,
                            const float* __restrict__ b2, const int* __restrict__ batch,
                            float* __restrict__ sums, float* __restrict__ cnts) {
  int idx = blockIdx.x * blockDim.x + threadIdx.x;
  if (idx >= NN * 128) return;
  int n = idx >> 7, cch = idx & 127;
  float v = out2[idx] + b2[cch];
  v = v > 0.f ? v : 0.f;
  float h = x1[n * 512 + cch] + v;
  int g = batch[n];
  atomicAdd(&sums[g * 128 + cch], h);
  if (cch == 0) atomicAdd(&cnts[g], 1.0f);
}

// pooled -> fc -> batchnorm(over 64 graphs) -> relu -> two heads. One block.
__global__ __launch_bounds__(256)
void k_final(const float* __restrict__ sums, const float* __restrict__ cnts,
             const float* __restrict__ Wfc, const float* __restrict__ bfc,
             const float* __restrict__ gamma, const float* __restrict__ beta,
             const float* __restrict__ Wheat, const float* __restrict__ bheat,
             const float* __restrict__ Wdehyd, const float* __restrict__ bdehyd,
             float* __restrict__ out) {
  __shared__ float z[NG * 32];
  __shared__ float mu[32], rstd[32];
  int t = threadIdx.x;
  for (int i = t; i < NG * 32; i += 256) {
    int g = i >> 5, j = i & 31;
    float cnt = cnts[g]; cnt = cnt > 1.f ? cnt : 1.f;
    const float* p = sums + g * 128;
    float s = 0.f;
    for (int cc = 0; cc < 128; ++cc) s += p[cc] * Wfc[cc * 32 + j];
    z[i] = bfc[j] + s / cnt;
  }
  __syncthreads();
  if (t < 32) {
    float m = 0.f;
    for (int g = 0; g < NG; ++g) m += z[g * 32 + t];
    m *= (1.0f / NG);
    float v = 0.f;
    for (int g = 0; g < NG; ++g) { float d = z[g * 32 + t] - m; v += d * d; }
    v *= (1.0f / NG);
    mu[t] = m; rstd[t] = rsqrtf(v + 1e-5f);
  }
  __syncthreads();
  for (int i = t; i < NG * 32; i += 256) {
    int j = i & 31;
    float zz = gamma[j] * (z[i] - mu[j]) * rstd[j] + beta[j];
    z[i] = zz > 0.f ? zz : 0.f;
  }
  __syncthreads();
  for (int i = t; i < NG * 3; i += 256) {
    int g = i / 3, k = i % 3;
    float a = bheat[k];
    for (int j = 0; j < 32; ++j) a += z[g * 32 + j] * Wheat[j * 3 + k];
    out[i] = a;
  }
  for (int i = t; i < NG * 2; i += 256) {
    int g = i / 2, k = i % 2;
    float a = bdehyd[k];
    for (int j = 0; j < 32; ++j) a += z[g * 32 + j] * Wdehyd[j * 2 + k];
    out[NG * 3 + i] = a;
  }
}

// ---------------- launcher ----------------

extern "C" void kernel_launch(void* const* d_in, const int* in_sizes, int n_in,
                              void* d_out, int out_size, void* d_ws, size_t ws_size,
                              hipStream_t stream) {
  const float* x        = (const float*)d_in[0];
  const int*   ei       = (const int*)  d_in[1];
  const int*   batch    = (const int*)  d_in[2];
  const float* W1       = (const float*)d_in[3];
  const float* att_s1   = (const float*)d_in[4];
  const float* att_d1   = (const float*)d_in[5];
  const float* b1       = (const float*)d_in[6];
  const float* W2       = (const float*)d_in[7];
  const float* att_s2   = (const float*)d_in[8];
  const float* att_d2   = (const float*)d_in[9];
  const float* b2       = (const float*)d_in[10];
  const float* Wfc      = (const float*)d_in[11];
  const float* bfc      = (const float*)d_in[12];
  const float* bn_g     = (const float*)d_in[13];
  const float* bn_b     = (const float*)d_in[14];
  const float* Wheat    = (const float*)d_in[15];
  const float* bheat    = (const float*)d_in[16];
  const float* Wdehyd   = (const float*)d_in[17];
  const float* bdehyd   = (const float*)d_in[18];
  float* out = (float*)d_out;

  // --- workspace arena (256B-aligned chunks) ---
  char* base = (char*)d_ws;
  size_t off = 0;
  auto alloc = [&](size_t bytes) -> char* {
    char* p = base + off;
    off = (off + bytes + 255) & ~(size_t)255;
    return p;
  };
  float*    h1   = (float*)   alloc((size_t)NN * 512 * 4);
  _Float16* x1h  = (_Float16*)alloc((size_t)NN * 512 * 2);
  _Float16* W2t  = (_Float16*)alloc((size_t)512 * 128 * 2);
  float*    h2   = (float*)   alloc((size_t)NN * 128 * 4);
  float*    a_s1 = (float*)   alloc((size_t)NN * 8 * 4);
  float*    a_d1 = (float*)   alloc((size_t)NN * 8 * 4);
  float*    a_s2 = (float*)   alloc((size_t)NN * 4 * 4);
  float*    a_d2 = (float*)   alloc((size_t)NN * 4 * 4);
  float*    e1   = (float*)   alloc((size_t)ET * 8 * 4);
  float*    e2   = (float*)   alloc((size_t)ET * 4 * 4);
  float*    mblk = (float*)   alloc((size_t)NN * 12 * 4);   // m1 (N*8) | m2 (N*4)
  float*    m1 = mblk;
  float*    m2 = mblk + (size_t)NN * 8;
  const size_t ZFL = (size_t)NN*512 + (size_t)NN*128 + (size_t)NN*8 + (size_t)NN*4
                   + (size_t)NG*128 + NG;
  float*    zblk = (float*)   alloc(ZFL * 4);               // single memset region
  float*    x1     = zblk;                       // accum -> relu'd layer-1 out
  float*    out2   = x1 + (size_t)NN * 512;      // layer-2 accum
  float*    denom1 = out2 + (size_t)NN * 128;
  float*    denom2 = denom1 + (size_t)NN * 8;
  float*    sums   = denom2 + (size_t)NN * 4;
  float*    cnts   = sums + (size_t)NG * 128;

  const int T = 256;
  auto blocks = [](long long n) { return (unsigned)((n + 255) / 256); };

  hipMemsetAsync(zblk, 0, ZFL * 4, stream);
  k_fill<<<blocks((long long)NN * 12), T, 0, stream>>>(mblk, -3.402823466e38f, NN * 12);

  // ---- layer 1 ----
  k_gemm1<<<blocks((long long)NN * 512), T, 0, stream>>>(x, W1, h1);
  k_attdot<<<blocks((long long)NN * 8), T, 0, stream>>>(h1, att_s1, att_d1, a_s1, a_d1, 8, 64);
  k_edge_logit<8><<<blocks((long long)ET * 8), T, 0, stream>>>(ei, a_s1, a_d1, e1, m1);
  k_edge_exp<8><<<blocks((long long)ET * 8), T, 0, stream>>>(ei, e1, m1, denom1);
  k_edge_scatter<8, 64><<<blocks((long long)ET * 128), T, 0, stream>>>(ei, e1, denom1, h1, x1);
  k_bias_relu_f16<<<blocks((long long)NN * 512), T, 0, stream>>>(x1, b1, x1h);

  // ---- layer 2 GEMM via WMMA ----
  k_w2t<<<blocks(512 * 128), T, 0, stream>>>(W2, W2t);
  k_gemm2_wmma<<<dim3((NN / 16) * 8 / 8), 256, 0, stream>>>(x1h, W2t, h2);  // 1875 blocks

  // ---- layer 2 attention ----
  k_attdot<<<blocks((long long)NN * 4), T, 0, stream>>>(h2, att_s2, att_d2, a_s2, a_d2, 4, 32);
  k_edge_logit<4><<<blocks((long long)ET * 4), T, 0, stream>>>(ei, a_s2, a_d2, e2, m2);
  k_edge_exp<4><<<blocks((long long)ET * 4), T, 0, stream>>>(ei, e2, m2, denom2);
  k_edge_scatter<4, 32><<<blocks((long long)ET * 32), T, 0, stream>>>(ei, e2, denom2, h2, out2);

  // ---- skip + pool ----
  k_skip_pool<<<blocks((long long)NN * 128), T, 0, stream>>>(x1, out2, b2, batch, sums, cnts);

  // ---- head ----
  k_final<<<1, 256, 0, stream>>>(sums, cnts, Wfc, bfc, bn_g, bn_b,
                                 Wheat, bheat, Wdehyd, bdehyd, out);

  (void)in_sizes; (void)n_in; (void)out_size; (void)ws_size;
}
